// graph_cage_33930241638803
// MI455X (gfx1250) — matrix operand order repaired and checked by reading, not compiled, not run on hardware
//
#include <hip/hip_runtime.h>

// ---------------------------------------------------------------------------
// CDNA5 (gfx1250) wave32 WMMA implementation of the molecular-graph GNN.
// GEMMs use v_wmma_f32_16x16x32_f16 (f16 in, f32 accumulate); everything else
// (relu, row-normalize, segment sum, output head) stays in f32.
// ---------------------------------------------------------------------------

typedef __attribute__((ext_vector_type(16))) _Float16 v16h;
typedef __attribute__((ext_vector_type(8)))  _Float16 v8h;
typedef __attribute__((ext_vector_type(8)))  float    v8f;

#define B_MOL  128
#define NATOMS 64
#define NTOT   (B_MOL * NATOMS)   // 8192
#define DIM    128
#define DEPTH  3

// A-fragment (16x32 f16, M x K). ISA layout: lanes 0-15 = rows, K 0-7 & 16-23;
// lanes 16-31 = same rows, K 8-15 & 24-31.  'base' points at row 0 of the
// 16-row block, 'stride' in halves.  Two aligned 16B LDS loads per fragment.
__device__ __forceinline__ v16h load_a_frag(const _Float16* base, int stride,
                                            int k0, int lane) {
  int r   = lane & 15;
  int off = (lane >> 4) << 3;  // 0 or 8
  const _Float16* p = base + r * stride + k0 + off;
  v8h lo = *(const v8h*)(p);
  v8h hi = *(const v8h*)(p + 16);
  return __builtin_shufflevector(lo, hi, 0, 1, 2, 3, 4, 5, 6, 7,
                                 8, 9, 10, 11, 12, 13, 14, 15);
}

// B-fragment (32x16 f16, K x N). ISA layout: lanes 0-15 = cols, K 0-7 & 16-23;
// lanes 16-31 = cols, K 8-15 & 24-31.  Weights / h are stored TRANSPOSED in LDS
// ([n][k]) so this is two aligned 16B loads.
__device__ __forceinline__ v16h load_b_frag(const _Float16* baseT, int stride,
                                            int k0, int lane) {
  int n   = lane & 15;
  int off = (lane >> 4) << 3;  // 0 or 8
  const _Float16* p = baseT + n * stride + k0 + off;
  v8h lo = *(const v8h*)(p);
  v8h hi = *(const v8h*)(p + 16);
  return __builtin_shufflevector(lo, hi, 0, 1, 2, 3, 4, 5, 6, 7,
                                 8, 9, 10, 11, 12, 13, 14, 15);
}

__device__ __forceinline__ v8f wmma_f16(v16h a, v16h b, v8f c) {
  return __builtin_amdgcn_wmma_f32_16x16x32_f16(false, a, false, b,
                                                (short)0, c, false, false);
}

// ---------------------------------------------------------------------------
// Kernel 1: one workgroup per (molecule, graph).  4 waves; wave w owns output
// rows [16w, 16w+16).  Dynamic LDS layout (bytes):
//   Xh   [64][128]  f16   0      .. 16384   activations (row-major)
//   HT   [128][64]  f16   16384  .. 32768   h transposed ([n][k]) for adj GEMM
//   Wt   [128][128] f16   32768  .. 65536   weight transposed ([n][k])
//   adjh [64][64]   f16   65536  .. 73728   block-diagonal adjacency tile
//   msum [128]      f32   73728  .. 74240   molecule fingerprint accumulator
// ---------------------------------------------------------------------------
__global__ void __launch_bounds__(128) gnn_kernel(
    const float* __restrict__ atoms1, const float* __restrict__ adj1,
    const float* __restrict__ atoms2, const float* __restrict__ adj2,
    const float* __restrict__ Wp, const float* __restrict__ bp,
    const float* __restrict__ Wg, const float* __restrict__ bg,
    float* __restrict__ mout) {
  extern __shared__ char smem[];
  _Float16* Xh   = (_Float16*)(smem);
  _Float16* HT   = (_Float16*)(smem + 16384);
  _Float16* Wt   = (_Float16*)(smem + 32768);
  _Float16* adjh = (_Float16*)(smem + 65536);
  float*    msum = (float*)(smem + 73728);

  const int b    = blockIdx.x;
  const int g    = blockIdx.y;
  const int tid  = threadIdx.x;
  const int lane = tid & 31;          // wave32
  const int w    = tid >> 5;          // wave id 0..3
  const int row0 = b * NATOMS;

  const float* atoms = g ? atoms2 : atoms1;
  const float* adj   = g ? adj2 : adj1;

  // ---- cooperative loads: only the DIAGONAL 64x64 adjacency block ----------
  for (int i = tid; i < NATOMS * NATOMS; i += 128) {
    int m = i >> 6, k = i & 63;
    adjh[i] = (_Float16)adj[(size_t)(row0 + m) * NTOT + (row0 + k)];
  }
  for (int i = tid; i < NATOMS * DIM; i += 128)
    Xh[i] = (_Float16)atoms[(size_t)row0 * DIM + i];
  for (int i = tid; i < DIM * DIM; i += 128) {  // Wp[k][n] -> Wt[n][k]
    int k = i >> 7, n = i & 127;
    Wt[n * DIM + k] = (_Float16)Wp[i];
  }
  if (tid < DIM) msum[tid] = 0.0f;
  __syncthreads();

  const _Float16* Arow   = Xh + (w * 16) * DIM;      // wave's 16 activation rows
  const _Float16* AdjRow = adjh + (w * 16) * NATOMS; // wave's 16 adjacency rows
  const int hi8 = (lane >> 4) * 8;
  const int c0  = lane & 15;

  v8f acc[8];

  // ---- projection: v = atoms @ Wp + bp -------------------------------------
#pragma unroll
  for (int t = 0; t < 8; ++t) {
    float bv = bp[t * 16 + c0];
#pragma unroll
    for (int r = 0; r < 8; ++r) acc[t][r] = bv;
  }
#pragma unroll
  for (int kc = 0; kc < 4; ++kc) {
    v16h a = load_a_frag(Arow, DIM, kc * 32, lane);
#pragma unroll
    for (int t = 0; t < 8; ++t) {
      v16h bf = load_b_frag(Wt + (t * 16) * DIM, DIM, kc * 32, lane);
      acc[t] = wmma_f16(a, bf, acc[t]);
    }
  }
  __syncthreads();
#pragma unroll
  for (int t = 0; t < 8; ++t)
#pragma unroll
    for (int r = 0; r < 8; ++r)
      Xh[(w * 16 + r + hi8) * DIM + t * 16 + c0] = (_Float16)acc[t][r];

  // ---- DEPTH message-passing layers ----------------------------------------
  for (int l = 0; l < DEPTH; ++l) {
    __syncthreads();  // previous GEMM done reading Wt
    for (int i = tid; i < DIM * DIM; i += 128) {
      int k = i >> 7, n = i & 127;
      Wt[n * DIM + k] = (_Float16)Wg[l * DIM * DIM + i];
    }
    if (l + 1 < DEPTH)  // global_prefetch_b8 of next layer's weights
      __builtin_prefetch(Wg + (size_t)(l + 1) * DIM * DIM + tid * 128, 0, 3);
    __syncthreads();

    // h = relu(v @ Wg[l] + bg[l])
#pragma unroll
    for (int t = 0; t < 8; ++t) {
      float bv = bg[l * DIM + t * 16 + c0];
#pragma unroll
      for (int r = 0; r < 8; ++r) acc[t][r] = bv;
    }
#pragma unroll
    for (int kc = 0; kc < 4; ++kc) {
      v16h a = load_a_frag(Arow, DIM, kc * 32, lane);
#pragma unroll
      for (int t = 0; t < 8; ++t) {
        v16h bf = load_b_frag(Wt + (t * 16) * DIM, DIM, kc * 32, lane);
        acc[t] = wmma_f16(a, bf, acc[t]);
      }
    }
    // relu in f32; stash h transposed into HT[n][k] for the adjacency GEMM
#pragma unroll
    for (int t = 0; t < 8; ++t)
#pragma unroll
      for (int r = 0; r < 8; ++r) {
        float h = fmaxf(acc[t][r], 0.0f);
        acc[t][r] = h;  // accumulator seeds hs = h + adj @ h
        HT[(t * 16 + c0) * NATOMS + (w * 16 + r + hi8)] = (_Float16)h;
      }
    __syncthreads();  // HT complete

    // hs = h + adjBlock @ h   (K = 64)
#pragma unroll
    for (int kc = 0; kc < 2; ++kc) {
      v16h a = load_a_frag(AdjRow, NATOMS, kc * 32, lane);
#pragma unroll
      for (int t = 0; t < 8; ++t) {
        v16h bf = load_b_frag(HT + (t * 16) * NATOMS, NATOMS, kc * 32, lane);
        acc[t] = wmma_f16(a, bf, acc[t]);
      }
    }

    // row L2 normalize: each wave holds its rows' full 128 columns.
    // Reduce across 16-lane halves with xor butterflies -- no LDS traffic.
#pragma unroll
    for (int r = 0; r < 8; ++r) {
      float ss = 0.0f;
#pragma unroll
      for (int t = 0; t < 8; ++t) ss += acc[t][r] * acc[t][r];
      ss += __shfl_xor(ss, 1);
      ss += __shfl_xor(ss, 2);
      ss += __shfl_xor(ss, 4);
      ss += __shfl_xor(ss, 8);
      float scale = 1.0f / fmaxf(sqrtf(ss), 1e-12f);
#pragma unroll
      for (int t = 0; t < 8; ++t) acc[t][r] *= scale;
    }

    if (l + 1 < DEPTH) {
#pragma unroll
      for (int t = 0; t < 8; ++t)
#pragma unroll
        for (int r = 0; r < 8; ++r)
          Xh[(w * 16 + r + hi8) * DIM + t * 16 + c0] = (_Float16)acc[t][r];
    } else {
      // segment sum over the molecule's 64 atoms (f32 LDS atomics, once)
#pragma unroll
      for (int t = 0; t < 8; ++t) {
        float cs = 0.0f;
#pragma unroll
        for (int r = 0; r < 8; ++r) cs += acc[t][r];
        atomicAdd(&msum[t * 16 + c0], cs);
      }
    }
  }
  __syncthreads();
  if (tid < DIM) mout[((size_t)g * B_MOL + b) * DIM + tid] = msum[tid];
}

// ---------------------------------------------------------------------------
// Kernel 2: merge + MLP + output head.  One workgroup, 8 waves; wave w owns
// output rows [16w, 16w+16).  concat(m1,m2) @ Wm done as two K=128 passes so
// LDS stays at 64 KB (Xh [128][128] f16 + Wt [128][128] f16).
// ---------------------------------------------------------------------------
__global__ void __launch_bounds__(256) merge_kernel(
    const float* __restrict__ mws,   // [2][128][128]
    const float* __restrict__ Wm,    // [256][128]
    const float* __restrict__ Wmlp,  // [2][128][128]
    const float* __restrict__ bmlp,  // [2][128]
    const float* __restrict__ Wo,    // [128][2]
    const float* __restrict__ bo,    // [2]
    float* __restrict__ out) {       // [128][2]
  extern __shared__ char smem[];
  _Float16* Xh = (_Float16*)(smem);           // [128][128]
  _Float16* Wt = (_Float16*)(smem + 32768);   // [128][128] transposed [n][k]

  const int tid  = threadIdx.x;
  const int lane = tid & 31;
  const int w    = tid >> 5;  // 0..7
  const int hi8  = (lane >> 4) * 8;
  const int c0   = lane & 15;

  v8f acc[8];
#pragma unroll
  for (int t = 0; t < 8; ++t)
#pragma unroll
    for (int r = 0; r < 8; ++r) acc[t][r] = 0.0f;

  // merged = [m1 | m2] @ Wm  (no bias), two K=128 passes
  for (int half = 0; half < 2; ++half) {
    __syncthreads();
    for (int i = tid; i < DIM * DIM; i += 256)
      Xh[i] = (_Float16)mws[half * DIM * DIM + i];
    for (int i = tid; i < DIM * DIM; i += 256) {
      int k = i >> 7, n = i & 127;
      Wt[n * DIM + k] = (_Float16)Wm[(half * DIM + k) * DIM + n];
    }
    __syncthreads();
    const _Float16* Arow = Xh + (w * 16) * DIM;
#pragma unroll
    for (int kc = 0; kc < 4; ++kc) {
      v16h a = load_a_frag(Arow, DIM, kc * 32, lane);
#pragma unroll
      for (int t = 0; t < 8; ++t) {
        v16h bf = load_b_frag(Wt + (t * 16) * DIM, DIM, kc * 32, lane);
        acc[t] = wmma_f16(a, bf, acc[t]);
      }
    }
  }
  __syncthreads();
#pragma unroll
  for (int t = 0; t < 8; ++t)
#pragma unroll
    for (int r = 0; r < 8; ++r)
      Xh[(w * 16 + r + hi8) * DIM + t * 16 + c0] = (_Float16)acc[t][r];

  // two MLP layers with relu (dropout p=0 -> identity)
  for (int i = 0; i < 2; ++i) {
    __syncthreads();
    for (int j = tid; j < DIM * DIM; j += 256) {
      int k = j >> 7, n = j & 127;
      Wt[n * DIM + k] = (_Float16)Wmlp[i * DIM * DIM + j];
    }
    __syncthreads();
#pragma unroll
    for (int t = 0; t < 8; ++t) {
      float bv = bmlp[i * DIM + t * 16 + c0];
#pragma unroll
      for (int r = 0; r < 8; ++r) acc[t][r] = bv;
    }
    const _Float16* Arow = Xh + (w * 16) * DIM;
#pragma unroll
    for (int kc = 0; kc < 4; ++kc) {
      v16h a = load_a_frag(Arow, DIM, kc * 32, lane);
#pragma unroll
      for (int t = 0; t < 8; ++t) {
        v16h bf = load_b_frag(Wt + (t * 16) * DIM, DIM, kc * 32, lane);
        acc[t] = wmma_f16(a, bf, acc[t]);
      }
    }
    __syncthreads();
#pragma unroll
    for (int t = 0; t < 8; ++t)
#pragma unroll
      for (int r = 0; r < 8; ++r)
        Xh[(w * 16 + r + hi8) * DIM + t * 16 + c0] =
            (_Float16)fmaxf(acc[t][r], 0.0f);
  }
  __syncthreads();

  // output head: [128,128] @ [128,2] + bo  -> 256 scalar dot products
  {
    int bb = tid >> 1, j = tid & 1;
    float s = bo[j];
    for (int k = 0; k < DIM; ++k)
      s += (float)Xh[bb * DIM + k] * Wo[k * 2 + j];
    out[bb * 2 + j] = s;
  }
}

// ---------------------------------------------------------------------------
extern "C" void kernel_launch(void* const* d_in, const int* in_sizes, int n_in,
                              void* d_out, int out_size, void* d_ws,
                              size_t ws_size, hipStream_t stream) {
  (void)in_sizes; (void)n_in; (void)out_size; (void)ws_size;
  const float* atoms1 = (const float*)d_in[0];
  const float* adj1   = (const float*)d_in[1];
  const float* atoms2 = (const float*)d_in[2];
  const float* adj2   = (const float*)d_in[3];
  const float* Wp     = (const float*)d_in[4];
  const float* bp     = (const float*)d_in[5];
  const float* Wg     = (const float*)d_in[6];
  const float* bg     = (const float*)d_in[7];
  const float* Wm     = (const float*)d_in[8];
  const float* Wmlp   = (const float*)d_in[9];
  const float* bmlp   = (const float*)d_in[10];
  const float* Wo     = (const float*)d_in[11];
  const float* bo     = (const float*)d_in[12];
  float* mws = (float*)d_ws;  // [2][128][128] molecule fingerprints

  const size_t lds1 = 74240;  // > 64KB static default, so request dynamically
  const size_t lds2 = 65536;
  (void)hipFuncSetAttribute((const void*)gnn_kernel,
                            hipFuncAttributeMaxDynamicSharedMemorySize,
                            (int)lds1);
  (void)hipFuncSetAttribute((const void*)merge_kernel,
                            hipFuncAttributeMaxDynamicSharedMemorySize,
                            (int)lds2);

  gnn_kernel<<<dim3(B_MOL, 2), 128, lds1, stream>>>(
      atoms1, adj1, atoms2, adj2, Wp, bp, Wg, bg, mws);
  merge_kernel<<<dim3(1), 256, lds2, stream>>>(mws, Wm, Wmlp, bmlp, Wo, bo,
                                               (float*)d_out);
}
